// Attention_53128745452063
// MI455X (gfx1250) — matrix operand order repaired and checked by reading, not compiled, not run on hardware
//
#include <hip/hip_runtime.h>

#define B_ 64
#define S_ 2048
#define H_ 256
#define BS_ (B_ * S_)

typedef __attribute__((ext_vector_type(16))) __bf16 v16bf;
typedef __attribute__((ext_vector_type(8)))  float  v8f;
typedef int v4i __attribute__((vector_size(16)));   // matches builtin's int4 param

// Branch-free tanh: native gfx1250 v_tanh_f32 if available, else exp2-based.
__device__ __forceinline__ float fast_tanh(float x) {
#if __has_builtin(__builtin_amdgcn_tanhf)
    return __builtin_amdgcn_tanhf(x);
#else
    float ax = __builtin_fabsf(x);
    float t  = __expf(-2.0f * ax);
    float r  = (1.0f - t) * __builtin_amdgcn_rcpf(1.0f + t);
    return __builtin_copysignf(r, x);
#endif
}

// ---------------- W_enc f32 -> bf16 ----------------
__global__ __launch_bounds__(256) void k_cvt_w(const float* __restrict__ W,
                                               __bf16* __restrict__ Wb, int n) {
    int i = blockIdx.x * 256 + threadIdx.x;
    if (i < n) Wb[i] = (__bf16)W[i];
}

// ---------------- dec_t[b][o] = sum_k dh[b][k] * W_dec[o][k] ----------------
__global__ __launch_bounds__(256) void k_dect(const float* __restrict__ dh,
                                              const float* __restrict__ Wdec,
                                              float* __restrict__ dect) {
    __shared__ float sdh[H_];
    const int b = blockIdx.x, o = threadIdx.x;
    sdh[o] = dh[b * H_ + o];
    __syncthreads();
    const float4* wr = (const float4*)(Wdec + (size_t)o * H_);
    float acc = 0.f;
    #pragma unroll 8
    for (int k4 = 0; k4 < H_ / 4; ++k4) {
        float4 w = wr[k4];
        acc += w.x * sdh[k4 * 4 + 0] + w.y * sdh[k4 * 4 + 1] +
               w.z * sdh[k4 * 4 + 2] + w.w * sdh[k4 * 4 + 3];
    }
    dect[b * H_ + o] = acc;
}

// ---------------- fused enc@W_enc^T + tanh + w_align dot (WMMA bf16) --------
#define WROWS  64
#define WPITCH 264   // 256 + 8 pad: 4-bank advance per row -> spread banks

__global__ __launch_bounds__(256)
void k_scores(const float* __restrict__ enc,
              const __bf16* __restrict__ Wb,
              const float* __restrict__ dect,
              const float* __restrict__ walign,
              float* __restrict__ raw) {
    __shared__ __bf16 sW[WROWS * WPITCH];
    __shared__ float  sDec[H_];
    __shared__ float  sWal[H_];

    const int tid   = threadIdx.x;
    const int lane  = tid & 31;
    const int wave  = tid >> 5;
    const int b     = blockIdx.y;
    const int sbase = blockIdx.x * 128 + wave * 16;

    sDec[tid] = dect[b * H_ + tid];
    sWal[tid] = walign[tid];

    const int half = lane >> 4;
    const int mrow = lane & 15;
    const float* rowp = enc + ((size_t)(b * S_ + sbase + mrow)) * H_;

    // A fragments: 16 rows x K=256, bf16, ISA 16-bit A 16x32 layout.
    v16bf a[8];
    #pragma unroll
    for (int kt = 0; kt < 8; ++kt) {
        const float* p0 = rowp + kt * 32 + half * 8;
        float4 f0 = *(const float4*)(p0);
        float4 f1 = *(const float4*)(p0 + 4);
        float4 f2 = *(const float4*)(p0 + 16);
        float4 f3 = *(const float4*)(p0 + 20);
        union { v16bf v; __bf16 e[16]; } ua;
        ua.e[0]=(__bf16)f0.x;  ua.e[1]=(__bf16)f0.y;  ua.e[2]=(__bf16)f0.z;  ua.e[3]=(__bf16)f0.w;
        ua.e[4]=(__bf16)f1.x;  ua.e[5]=(__bf16)f1.y;  ua.e[6]=(__bf16)f1.z;  ua.e[7]=(__bf16)f1.w;
        ua.e[8]=(__bf16)f2.x;  ua.e[9]=(__bf16)f2.y;  ua.e[10]=(__bf16)f2.z; ua.e[11]=(__bf16)f2.w;
        ua.e[12]=(__bf16)f3.x; ua.e[13]=(__bf16)f3.y; ua.e[14]=(__bf16)f3.z; ua.e[15]=(__bf16)f3.w;
        a[kt] = ua.v;
    }

    float partial[8] = {0.f,0.f,0.f,0.f,0.f,0.f,0.f,0.f};
    const int col = mrow;

    for (int c = 0; c < 4; ++c) {          // 4 chunks of 64 W rows
        __syncthreads();
#if __has_builtin(__builtin_amdgcn_global_load_async_to_lds_b128)
        // CDNA5 async DMA: global -> LDS with no VGPR round-trip (ASYNCcnt).
        #pragma unroll
        for (int j = 0; j < 8; ++j) {
            int linear = tid + j * 256;
            int row    = linear >> 5;
            int c16    = linear & 31;
            uintptr_t gsrc  = (uintptr_t)((const uint4*)(Wb + (size_t)(c * WROWS + row) * H_) + c16);
            unsigned  ldsoff = (unsigned)(uintptr_t)(sW + row * WPITCH) + (unsigned)(c16 * 16);
            __builtin_amdgcn_global_load_async_to_lds_b128(
                (__attribute__((address_space(1))) v4i*)gsrc,
                (__attribute__((address_space(3))) v4i*)(uintptr_t)ldsoff,
                0, 0);
        }
#if __has_builtin(__builtin_amdgcn_s_wait_asynccnt)
        __builtin_amdgcn_s_wait_asynccnt(0);
#else
        asm volatile("s_wait_asynccnt 0x0" ::: "memory");
#endif
#else
        // Fallback: two-phase copy so the 8 loads pipeline on LOADcnt.
        uint4 tmp[8];
        #pragma unroll
        for (int j = 0; j < 8; ++j) {
            int linear = tid + j * 256;
            int row    = linear >> 5;
            int c16    = linear & 31;
            tmp[j] = *((const uint4*)(Wb + (size_t)(c * WROWS + row) * H_) + c16);
        }
        #pragma unroll
        for (int j = 0; j < 8; ++j) {
            int linear = tid + j * 256;
            int row    = linear >> 5;
            int c16    = linear & 31;
            *((uint4*)(sW + row * WPITCH) + c16) = tmp[j];
        }
#endif
        __syncthreads();

        #pragma unroll
        for (int ntl = 0; ntl < 4; ++ntl) {
            const int nt = c * 4 + ntl;
            v8f acc = {0.f,0.f,0.f,0.f,0.f,0.f,0.f,0.f};
            const __bf16* wrow = sW + (ntl * 16 + col) * WPITCH + half * 8;

            union { v16bf v; uint4 q[2]; } ub[2];
            ub[0].q[0] = ((const uint4*)(wrow))[0];
            ub[0].q[1] = ((const uint4*)(wrow))[2];
            #pragma unroll
            for (int kt = 0; kt < 8; ++kt) {
                if (kt < 7) {   // prefetch next B fragment while WMMA kt executes
                    ub[(kt + 1) & 1].q[0] = ((const uint4*)(wrow + (kt + 1) * 32))[0];
                    ub[(kt + 1) & 1].q[1] = ((const uint4*)(wrow + (kt + 1) * 32))[2];
                }
                acc = __builtin_amdgcn_wmma_f32_16x16x32_bf16(
                        false, a[kt], false, ub[kt & 1].v, (short)0, acc, false, false);
            }
            const int   ng   = nt * 16 + col;
            const float bias = sDec[ng];
            const float wal  = sWal[ng];
            #pragma unroll
            for (int i = 0; i < 8; ++i)      // VGPR i = row (i or 8+i), lane = col
                partial[i] += fast_tanh(acc[i] + bias) * wal;
        }
    }

    // reduce over 16 columns (each 16-lane half independently)
    #pragma unroll
    for (int i = 0; i < 8; ++i) {
        float v = partial[i];
        v += __shfl_xor(v, 1, 32);
        v += __shfl_xor(v, 2, 32);
        v += __shfl_xor(v, 4, 32);
        v += __shfl_xor(v, 8, 32);
        partial[i] = v;
    }
    if ((lane & 15) == 0) {                  // lane0 -> rows 0..7, lane16 -> rows 8..15
        float* out = raw + (size_t)b * S_ + sbase + half * 8;
        #pragma unroll
        for (int i = 0; i < 8; ++i) out[i] = partial[i];
    }
}

// ---------------- global softmax over flattened B*S ----------------
__global__ __launch_bounds__(256) void k_red_max(const float* __restrict__ raw,
                                                 float* __restrict__ pmax) {
    __shared__ float s[256];
    int i = blockIdx.x * 512 + threadIdx.x;
    s[threadIdx.x] = fmaxf(raw[i], raw[i + 256]);
    __syncthreads();
    for (int o = 128; o > 0; o >>= 1) {
        if (threadIdx.x < o) s[threadIdx.x] = fmaxf(s[threadIdx.x], s[threadIdx.x + o]);
        __syncthreads();
    }
    if (threadIdx.x == 0) pmax[blockIdx.x] = s[0];
}

__global__ __launch_bounds__(256) void k_final_max(const float* __restrict__ pmax,
                                                   float* __restrict__ gmax) {
    __shared__ float s[256];
    s[threadIdx.x] = pmax[threadIdx.x];
    __syncthreads();
    for (int o = 128; o > 0; o >>= 1) {
        if (threadIdx.x < o) s[threadIdx.x] = fmaxf(s[threadIdx.x], s[threadIdx.x + o]);
        __syncthreads();
    }
    if (threadIdx.x == 0) gmax[0] = s[0];
}

__global__ __launch_bounds__(256) void k_exps(float* __restrict__ raw,
                                              const float* __restrict__ gmax,
                                              float* __restrict__ psum) {
    __shared__ float s[256];
    const float gm = gmax[0];
    int i = blockIdx.x * 512 + threadIdx.x;
    float e0 = __expf(raw[i] - gm);
    float e1 = __expf(raw[i + 256] - gm);
    raw[i] = e0; raw[i + 256] = e1;
    s[threadIdx.x] = e0 + e1;
    __syncthreads();
    for (int o = 128; o > 0; o >>= 1) {
        if (threadIdx.x < o) s[threadIdx.x] += s[threadIdx.x + o];
        __syncthreads();
    }
    if (threadIdx.x == 0) psum[blockIdx.x] = s[0];
}

__global__ __launch_bounds__(256) void k_final_sum(const float* __restrict__ psum,
                                                   float* __restrict__ gsum) {
    __shared__ float s[256];
    s[threadIdx.x] = psum[threadIdx.x];
    __syncthreads();
    for (int o = 128; o > 0; o >>= 1) {
        if (threadIdx.x < o) s[threadIdx.x] += s[threadIdx.x + o];
        __syncthreads();
    }
    if (threadIdx.x == 0) gsum[0] = s[0];
}

__global__ __launch_bounds__(256) void k_norm(const float* __restrict__ raw,
                                              const float* __restrict__ gsum,
                                              float* __restrict__ scores) {
    const float inv = 1.0f / gsum[0];
    int i = blockIdx.x * 256 + threadIdx.x;
    scores[i] = raw[i] * inv;
}

// ---------------- context[b][h] = sum_s scores[b][s] * enc[b][s][h] ----------
__global__ __launch_bounds__(256) void k_context(const float* __restrict__ enc,
                                                 const float* __restrict__ scores,
                                                 float* __restrict__ ctx) {
    __shared__ float sp[256];
    const int b = blockIdx.x, h = threadIdx.x;
    float acc = 0.f;
    for (int s0 = 0; s0 < S_; s0 += 256) {
        __syncthreads();
        sp[h] = scores[(size_t)b * S_ + s0 + h];
        __syncthreads();
        const float* e = enc + ((size_t)(b * S_ + s0)) * H_ + h;
        #pragma unroll 8
        for (int j = 0; j < 256; ++j) acc += sp[j] * e[(size_t)j * H_];
    }
    ctx[b * H_ + h] = acc;
}

extern "C" void kernel_launch(void* const* d_in, const int* in_sizes, int n_in,
                              void* d_out, int out_size, void* d_ws, size_t ws_size,
                              hipStream_t stream) {
    const float* enc    = (const float*)d_in[0];   // [B,S,H]
    const float* dh     = (const float*)d_in[1];   // [B,H]
    const float* Wdec   = (const float*)d_in[2];   // [H,H]
    const float* Wenc   = (const float*)d_in[3];   // [H,H]
    const float* walign = (const float*)d_in[4];   // [1,H]

    char* w = (char*)d_ws;
    __bf16* Wb   = (__bf16*)(w);                 // 131072 B
    float*  dect = (float*)(w + 131072);          // 65536 B
    float*  raw  = (float*)(w + 196608);          // 524288 B
    float*  pmax = (float*)(w + 720896);          // 1024 B
    float*  psum = (float*)(w + 721920);          // 1024 B
    float*  gmax = (float*)(w + 722944);          // 4 B
    float*  gsum = (float*)(w + 722948);          // 4 B

    float* out_ctx    = (float*)d_out;            // [B,1,H]
    float* out_scores = (float*)d_out + B_ * H_;  // [B,S]

    k_cvt_w   <<<H_ * H_ / 256, 256, 0, stream>>>(Wenc, Wb, H_ * H_);
    k_dect    <<<B_, 256, 0, stream>>>(dh, Wdec, dect);
    k_scores  <<<dim3(S_ / 128, B_), 256, 0, stream>>>(enc, Wb, dect, walign, raw);
    k_red_max <<<BS_ / 512, 256, 0, stream>>>(raw, pmax);
    k_final_max<<<1, 256, 0, stream>>>(pmax, gmax);
    k_exps    <<<BS_ / 512, 256, 0, stream>>>(raw, gmax, psum);
    k_final_sum<<<1, 256, 0, stream>>>(psum, gsum);
    k_norm    <<<BS_ / 256, 256, 0, stream>>>(raw, gsum, out_scores);
    k_context <<<B_, 256, 0, stream>>>(enc, out_scores, out_ctx);
}
